// DialogueGNNModel_1769526526152
// MI455X (gfx1250) — compile-verified, matrix operand order checked by reading, not run on hardware
//
#include <hip/hip_runtime.h>
#include <math.h>

// Problem constants (match reference)
#define Nn 200000
#define Ee 400000
#define Hh 256
#define Rr 2
#define Cc 6
#define RT 4   // row-tiles (of 16 rows) per GEMM block

typedef __attribute__((ext_vector_type(16))) __bf16 v16bf;
typedef __attribute__((ext_vector_type(8)))  float  v8f;

union Frag { v16bf v; unsigned int u[8]; };

// round-to-nearest-even f32 -> bf16, packed pair into one dword
__device__ __forceinline__ unsigned int pk2bf(float a, float b) {
    unsigned int ua = __float_as_uint(a), ub = __float_as_uint(b);
    unsigned int ra = (ua + 0x7FFFu + ((ua >> 16) & 1u)) >> 16;
    unsigned int rb = (ub + 0x7FFFu + ((ub >> 16) & 1u)) >> 16;
    return (ra & 0xFFFFu) | ((rb & 0xFFFFu) << 16);
}

// map (lane, j) -> K index within a 32-wide chunk (A and B share this packing)
__device__ __forceinline__ int frag_kk(int lane, int j) {
    int kbase = (lane >> 4) * 8;
    return (j < 4) ? (kbase + 2 * j) : (16 + kbase + 2 * (j - 4));
}

// ---------------------------------------------------------------------------
// Pack a K x 256 (col-tiled) B matrix into WMMA bf16 B-fragment layout.
// Storage: dword index = ((ct*KC + kc)*8 + j)*32 + lane  (j-major-then-lane so
// the GEMM's per-j fragment load is a fully coalesced 128B global_load_b32).
// Lane layout (16-bit B, 32x16 K x N): n = lane&15; lanes<16 hold K {0..7,16..23},
// lanes>=16 hold K {8..15,24..31}; vgpr j holds the K pair (kk, kk+1).
// transpose==0: B[k][n] = W[k*ldw + n]   (w1/w2: einsum 'nh,rho->rno')
// transpose==1: B[k][n] = W[n*ldw + k]   (lin_W is [H, 2H], we need lin_W^T)
// ---------------------------------------------------------------------------
__global__ void __launch_bounds__(256)
pack_b_kernel(const float* __restrict__ W, unsigned int* __restrict__ P,
              int KC, int ldw, int transpose) {
    int tid  = blockIdx.x * 256 + threadIdx.x;
    int lane = tid & 31;
    int j    = (tid >> 5) & 7;
    int ckc  = tid >> 8;             // == blockIdx.x ; one block per (ct,kc)
    int kc   = ckc % KC;
    int ct   = ckc / KC;
    if (ct >= 16) return;

    int n = ct * 16 + (lane & 15);
    int k = kc * 32 + frag_kk(lane, j);

    float v0 = transpose ? W[(size_t)n * ldw + k]       : W[(size_t)k * ldw + n];
    float v1 = transpose ? W[(size_t)n * ldw + (k + 1)] : W[(size_t)(k + 1) * ldw + n];
    P[tid] = pk2bf(v0, v1);
}

// ---------------------------------------------------------------------------
// Y[64 rows x 256 cols] = X[64 x 256] @ Bpack (256x256), bf16 WMMA, f32 accum.
// 256 threads = 8 waves.  Phase 1: all threads cooperatively convert the 64-row
// x-tile into bf16 A fragments in LDS (exact per-lane WMMA layout) -- done ONCE
// per block, outside the K loop.  Phase 2: wave w owns col-tiles 2w and 2w+1,
// processed as two sequential halves so only ONE tile's B fragments (64 VGPRs)
// are resident at a time -> no scratch spills.  Inner loop per kc is just
// 2x ds_load_b128 (A) + v_wmma.  64 WMMAs per wave per block.
// ---------------------------------------------------------------------------
__global__ void __launch_bounds__(256)
gemm256_wmma(const float* __restrict__ X, const unsigned int* __restrict__ P,
             float* __restrict__ Y) {
    __shared__ unsigned int af[RT * 8 * 32 * 8];   // 32 KB: [rt][kc][lane][j]
    const int tid  = threadIdx.x;
    const int lane = tid & 31;
    const int wave = tid >> 5;
    const int row0 = blockIdx.x * (RT * 16);

    // Phase 1: stage bf16 A fragments
    for (int i = tid; i < RT * 8 * 32 * 8; i += 256) {
        int j  = i & 7;
        int ln = (i >> 3) & 31;
        int kc = (i >> 8) & 7;
        int rt = i >> 11;
        int m  = ln & 15;
        int k  = kc * 32 + frag_kk(ln, j);
        const float* xr = X + (size_t)(row0 + rt * 16 + m) * Hh + k;
        af[i] = pk2bf(xr[0], xr[1]);
    }
    __syncthreads();

    const int rrow = (lane >> 4) * 8;
    const int nl   = lane & 15;

    for (int half = 0; half < 2; ++half) {
        const int ct = 2 * wave + half;
        // B fragments for this col-tile: resident in VGPRs across all row-tiles
        Frag B[8];
        const unsigned int* p = P + (size_t)ct * 8 * 256;
#pragma unroll
        for (int kc = 0; kc < 8; ++kc) {
#pragma unroll
            for (int j = 0; j < 8; ++j)
                B[kc].u[j] = p[kc * 256 + j * 32 + lane];
        }

        for (int rt = 0; rt < RT; ++rt) {
            v8f acc = {0.f,0.f,0.f,0.f,0.f,0.f,0.f,0.f};
#pragma unroll
            for (int kc = 0; kc < 8; ++kc) {
                const uint4* ap = (const uint4*)&af[((rt * 8 + kc) * 32 + lane) * 8];
                uint4 lo = ap[0], hi = ap[1];
                Frag a;
                a.u[0] = lo.x; a.u[1] = lo.y; a.u[2] = lo.z; a.u[3] = lo.w;
                a.u[4] = hi.x; a.u[5] = hi.y; a.u[6] = hi.z; a.u[7] = hi.w;
                acc = __builtin_amdgcn_wmma_f32_16x16x32_bf16(false, a.v, false, B[kc].v,
                                                              (short)0, acc, false, false);
            }
            // D layout: vgpr i -> row = i + (lane>=16)*8, col = ct*16 + (lane&15)
            float* y = Y + (size_t)(row0 + rt * 16 + rrow) * Hh + ct * 16 + nl;
#pragma unroll
            for (int i = 0; i < 8; ++i)
                y[(size_t)i * Hh] = acc[i];
        }
    }
}

// ---------------------------------------------------------------------------
// Edge message + segment-sum: one wave per edge; float4 gathers from
// xw[edge_type][src], scaled by 2*edge_norm, atomic f32 adds into agg[dst]
// (agg ~= L2-resident: 205 MB vs 192 MB L2). cnt counts edges per dst.
// ---------------------------------------------------------------------------
__global__ void __launch_bounds__(256)
edge_scatter(const int* __restrict__ ei, const float* __restrict__ en,
             const int* __restrict__ et, const float* __restrict__ XW,
             float* __restrict__ agg, float* __restrict__ cnt) {
    int e = blockIdx.x * 8 + (threadIdx.x >> 5);
    if (e >= Ee) return;
    int   lane = threadIdx.x & 31;
    int   d = ei[e];
    int   s = ei[Ee + e];
    int   t = et[e];
    float nrm = en[e] * 2.0f;
    const float* srcp = XW + ((size_t)t * Nn + s) * Hh;
    float*       dstp = agg + (size_t)d * Hh;
#pragma unroll
    for (int i = 0; i < 2; ++i) {
        int h = i * 128 + lane * 4;
        float4 v = *(const float4*)(srcp + h);
        atomicAdd(dstp + h + 0, v.x * nrm);
        atomicAdd(dstp + h + 1, v.y * nrm);
        atomicAdd(dstp + h + 2, v.z * nrm);
        atomicAdd(dstp + h + 3, v.w * nrm);
    }
    if (lane == 0) atomicAdd(cnt + d, 1.0f);
}

// mean + optional sigmoid
__global__ void __launch_bounds__(256)
finalize_mean(const float* __restrict__ agg, const float* __restrict__ cnt,
              float* __restrict__ out, int do_sigmoid) {
    size_t i = (size_t)blockIdx.x * 256 + threadIdx.x;
    if (i >= (size_t)Nn * Hh) return;
    int   n = (int)(i / Hh);
    float v = agg[i] / fmaxf(cnt[n], 1.0f);
    if (do_sigmoid) v = 1.0f / (1.0f + expf(-v));
    out[i] = v;
}

// ---------------------------------------------------------------------------
// Final head: hidden = relu([x | out2] @ linpack + lin_b) (WMMA, K=512),
// A fragments pre-staged in LDS (bf16), hidden kept in LDS, then fc (256->6)
// + log_softmax -> d_out directly.  Col-tiles processed as two halves to keep
// VGPR pressure low.
// ---------------------------------------------------------------------------
__global__ void __launch_bounds__(256)
final_fused(const float* __restrict__ X, const float* __restrict__ O2,
            const unsigned int* __restrict__ P,
            const float* __restrict__ linb,
            const float* __restrict__ fcW, const float* __restrict__ fcb,
            float* __restrict__ out) {
    __shared__ unsigned int af[16 * 32 * 8];       // 16 KB: [kc][lane][j], KC=16
    __shared__ float hid[16][Hh + 4];
    __shared__ float lg[16][8];

    const int tid  = threadIdx.x;
    const int lane = tid & 31;
    const int wave = tid >> 5;
    const int row0 = blockIdx.x * 16;
    const int KC   = 512 / 32;                      // 16

    // stage bf16 A fragments of [x | out2] (K = 512)
    for (int i = tid; i < 16 * 32 * 8; i += 256) {
        int j  = i & 7;
        int ln = (i >> 3) & 31;
        int kc = i >> 8;                            // 0..15
        int m  = ln & 15;
        int k  = kc * 32 + frag_kk(ln, j);          // k, k+1 stay on one side of 256
        size_t rbase = (size_t)(row0 + m) * Hh;
        float v0, v1;
        if (k < Hh) { v0 = X[rbase + k];        v1 = X[rbase + k + 1]; }
        else        { v0 = O2[rbase + (k - Hh)]; v1 = O2[rbase + (k - Hh) + 1]; }
        af[i] = pk2bf(v0, v1);
    }
    __syncthreads();

    const int rrow = (lane >> 4) * 8;
    const int nl   = lane & 15;

    for (int half = 0; half < 2; ++half) {
        const int ct = 2 * wave + half;
        const unsigned int* p = P + (size_t)ct * KC * 256;
        v8f acc = {0.f,0.f,0.f,0.f,0.f,0.f,0.f,0.f};
#pragma unroll 4
        for (int kc = 0; kc < KC; ++kc) {
            const uint4* ap = (const uint4*)&af[(kc * 32 + lane) * 8];
            uint4 lo = ap[0], hi = ap[1];
            Frag a;
            a.u[0] = lo.x; a.u[1] = lo.y; a.u[2] = lo.z; a.u[3] = lo.w;
            a.u[4] = hi.x; a.u[5] = hi.y; a.u[6] = hi.z; a.u[7] = hi.w;
            Frag b;
#pragma unroll
            for (int j = 0; j < 8; ++j)
                b.u[j] = p[kc * 256 + j * 32 + lane];
            acc = __builtin_amdgcn_wmma_f32_16x16x32_bf16(false, a.v, false, b.v,
                                                          (short)0, acc, false, false);
        }
        const float bb = linb[ct * 16 + nl];
#pragma unroll
        for (int i = 0; i < 8; ++i)
            hid[rrow + i][ct * 16 + nl] = fmaxf(acc[i] + bb, 0.0f);
    }
    __syncthreads();

    // fc: 16 rows x 6 classes = 96 dot products of length 256
    if (tid < 16 * Cc) {
        int r = tid / Cc, c = tid % Cc;
        float s = fcb[c];
        for (int k = 0; k < Hh; ++k) s += hid[r][k] * fcW[c * Hh + k];
        lg[r][c] = s;
    }
    __syncthreads();

    if (tid < 16) {
        int r = tid;
        float mx = lg[r][0];
#pragma unroll
        for (int c = 1; c < Cc; ++c) mx = fmaxf(mx, lg[r][c]);
        float se = 0.0f;
#pragma unroll
        for (int c = 0; c < Cc; ++c) se += expf(lg[r][c] - mx);
        float lse = mx + logf(se);
#pragma unroll
        for (int c = 0; c < Cc; ++c)
            out[(size_t)(row0 + r) * Cc + c] = lg[r][c] - lse;
    }
}

// ---------------------------------------------------------------------------
extern "C" void kernel_launch(void* const* d_in, const int* in_sizes, int n_in,
                              void* d_out, int out_size, void* d_ws, size_t ws_size,
                              hipStream_t stream) {
    const float* x    = (const float*)d_in[0];
    const int*   ei   = (const int*)  d_in[1];   // [2, E]
    const float* en   = (const float*)d_in[2];
    const int*   et   = (const int*)  d_in[3];
    const float* w1   = (const float*)d_in[4];   // [R, H, H]
    const float* w2   = (const float*)d_in[5];
    const float* linW = (const float*)d_in[6];   // [H, 2H]
    const float* linb = (const float*)d_in[7];
    const float* fcW  = (const float*)d_in[8];   // [C, H]
    const float* fcb  = (const float*)d_in[9];
    float* out = (float*)d_out;

    char* ws = (char*)d_ws;
    size_t off = 0;
    float* xw   = (float*)(ws + off); off += (size_t)Rr * Nn * Hh * 4;  // 409.6 MB
    float* agg  = (float*)(ws + off); off += (size_t)Nn * Hh * 4;       // 204.8 MB
    float* cntb = (float*)(ws + off); off += (size_t)Nn * 4;            // 0.8 MB
    float* obuf = (float*)(ws + off); off += (size_t)Nn * Hh * 4;       // 204.8 MB
    const size_t WPACK = 8 * 16 * 256;   // dwords per relation (KC=8)
    const size_t LPACK = 16 * 16 * 256;  // dwords for lin (KC=16)
    unsigned int* w1p  = (unsigned int*)(ws + off); off += Rr * WPACK * 4;
    unsigned int* w2p  = (unsigned int*)(ws + off); off += Rr * WPACK * 4;
    unsigned int* linp = (unsigned int*)(ws + off); off += LPACK * 4;
    (void)ws_size; (void)in_sizes; (void)n_in; (void)out_size;

    const int gblk = Nn / (16 * RT);     // 3125
    const int hblk = Nn / 16;            // 12500
    const int eblk = (Ee + 7) / 8;       // 50000
    const int fblk = (int)(((size_t)Nn * Hh + 255) / 256);

    // Pack weights into WMMA B-fragment layout (bf16).
    for (int r = 0; r < Rr; ++r) {
        pack_b_kernel<<<16 * 8, 256, 0, stream>>>(w1 + (size_t)r * Hh * Hh, w1p + r * WPACK, 8, Hh, 0);
        pack_b_kernel<<<16 * 8, 256, 0, stream>>>(w2 + (size_t)r * Hh * Hh, w2p + r * WPACK, 8, Hh, 0);
    }
    pack_b_kernel<<<16 * 16, 256, 0, stream>>>(linW, linp, 16, 2 * Hh, 1);

    // ---- Layer 1 ----
    hipMemsetAsync(agg, 0, (size_t)Nn * Hh * 4, stream);
    hipMemsetAsync(cntb, 0, (size_t)Nn * 4, stream);
    for (int r = 0; r < Rr; ++r)
        gemm256_wmma<<<gblk, 256, 0, stream>>>(x, w1p + r * WPACK, xw + (size_t)r * Nn * Hh);
    edge_scatter<<<eblk, 256, 0, stream>>>(ei, en, et, xw, agg, cntb);
    finalize_mean<<<fblk, 256, 0, stream>>>(agg, cntb, obuf, 1);

    // ---- Layer 2 ----
    hipMemsetAsync(agg, 0, (size_t)Nn * Hh * 4, stream);
    hipMemsetAsync(cntb, 0, (size_t)Nn * 4, stream);
    for (int r = 0; r < Rr; ++r)
        gemm256_wmma<<<gblk, 256, 0, stream>>>(obuf, w2p + r * WPACK, xw + (size_t)r * Nn * Hh);
    edge_scatter<<<eblk, 256, 0, stream>>>(ei, en, et, xw, agg, cntb);
    finalize_mean<<<fblk, 256, 0, stream>>>(agg, cntb, obuf, 0);

    // ---- Head: concat -> lin(+bias,relu) -> fc -> log_softmax ----
    final_fused<<<hblk, 256, 0, stream>>>(x, obuf, linp, linb, fcW, fcb, out);
}
